// ALiBiMultiHeadAttention_69552700391963
// MI455X (gfx1250) — matrix operand order repaired and checked by reading, not compiled, not run on hardware
//
#include <hip/hip_runtime.h>

// ---------------------------------------------------------------------------
// ALiBi multi-head attention, CDNA5 (gfx1250) WMMA implementation, round 4.
// B=2, S=2048, D=1024, H=16, DK=64. f16 WMMA, f32 accumulate, flash-attention.
// Round-4 change: ALiBi |i-j| carried as an f32 running base (one subtract
// per key chunk) instead of per-element int->float conversion, eliminating
// all v_cvt_f32_i32 from the attention hot loop.
// ---------------------------------------------------------------------------

typedef __attribute__((ext_vector_type(16))) _Float16 v16h;
typedef __attribute__((ext_vector_type(8)))  _Float16 v8h;
typedef __attribute__((ext_vector_type(8)))  float    v8f;

#define WMMA_F16(A, Bm, C) \
    __builtin_amdgcn_wmma_f32_16x16x32_f16(false, (A), false, (Bm), (short)0, (C), false, false)

static constexpr int Bb  = 2;
static constexpr int Ss  = 2048;
static constexpr int Dd  = 1024;
static constexpr int Hh  = 16;
static constexpr int DKk = 64;
static constexpr int Mtot = Bb * Ss;            // 4096 rows for the projections

union AFrag { v16h v; v8h h[2]; };

// -------------------------- f32 -> f16 convert -----------------------------
__global__ void __launch_bounds__(256)
cvt_f32_to_f16(const float* __restrict__ src, _Float16* __restrict__ dst, int n) {
    int i = blockIdx.x * blockDim.x + threadIdx.x;
    if (i < n) dst[i] = (_Float16)src[i];
}

// ------------------- weight transpose + convert: Wt[n][k] ------------------
__global__ void __launch_bounds__(256)
transpose_cvt(const float* __restrict__ W, _Float16* __restrict__ Wt) {
    int idx = blockIdx.x * blockDim.x + threadIdx.x;   // Dd*Dd threads
    int k = idx & (Dd - 1);
    int n = idx >> 10;
    Wt[(size_t)n * Dd + k] = (_Float16)W[(size_t)k * Dd + n];
}

// ---------- mask -> transposed additive f16 bias: mbT[b][j][i] -------------
// 0 where mask!=0, -60000 (acts as -inf through softmax) where mask==0.
// LDS-tiled 32x32 transpose: coalesced reads along j, coalesced writes along i.
__global__ void __launch_bounds__(256)
mask_bias_T(const int* __restrict__ mask, _Float16* __restrict__ mbT) {
    __shared__ _Float16 tile[32][33];
    const int b  = blockIdx.z;
    const int i0 = blockIdx.x * 32;
    const int j0 = blockIdx.y * 32;
    const int tx = threadIdx.x & 31;
    const int ty = threadIdx.x >> 5;               // 0..7
    const int* mrow = mask + (size_t)b * Ss * Ss;
#pragma unroll
    for (int k = 0; k < 4; ++k) {
        const int i = i0 + ty + 8 * k;
        tile[ty + 8 * k][tx] = (mrow[(size_t)i * Ss + j0 + tx] != 0)
                                   ? (_Float16)0.0f
                                   : (_Float16)(-60000.0f);
    }
    __syncthreads();
    _Float16* out = mbT + (size_t)b * Ss * Ss;
#pragma unroll
    for (int k = 0; k < 4; ++k) {
        const int j = j0 + ty + 8 * k;
        out[(size_t)j * Ss + i0 + tx] = tile[tx][ty + 8 * k];
    }
}

// ------------------------------ WMMA GEMM ----------------------------------
// Y[M=4096][N=1024] = X16 @ Wt^T + bias.  One wave computes a 32x64 tile:
// B-fragments feed both 16-row halves, A-fragments all four 16-col tiles.
// MODE 0: store f16 row-major [M][Dd]                   (Q, K)
// MODE 1: store f16 per-head transposed Vt[b][h][d][s]  (V)
// MODE 2: store f32 row-major [M][Dd] to d_out          (output projection)
template <int MODE>
__global__ void __launch_bounds__(128)
gemm_wmma(const _Float16* __restrict__ X, const _Float16* __restrict__ Wt,
          const float* __restrict__ bias, void* __restrict__ outp) {
    const int lane = threadIdx.x & 31;
    const int wid  = threadIdx.x >> 5;
    const int gw   = blockIdx.x * 4 + wid;       // wave id: 2048 total
    const int nGrp = Dd / 64;                    // 16 column groups
    const int mt   = gw / nGrp;                  // 0..127 (32-row tiles)
    const int ng   = gw % nGrp;                  // 0..15
    const int half = lane >> 4;
    const int l16  = lane & 15;

    const _Float16* xrow0 = X + (size_t)(mt * 32 + l16) * Dd;
    const _Float16* xrow1 = xrow0 + (size_t)16 * Dd;

    v8f acc[2][4] = {};
    for (int kk = 0; kk < Dd; kk += 32) {
        AFrag a0, a1;
        a0.h[0] = *(const v8h*)(xrow0 + kk + half * 16);
        a0.h[1] = *(const v8h*)(xrow0 + kk + half * 16 + 8);
        a1.h[0] = *(const v8h*)(xrow1 + kk + half * 16);
        a1.h[1] = *(const v8h*)(xrow1 + kk + half * 16 + 8);
#pragma unroll
        for (int t = 0; t < 4; ++t) {
            const int ncol = ng * 64 + t * 16 + l16;
            v16h bf = *(const v16h*)(Wt + (size_t)ncol * Dd + kk + half * 16);
            acc[0][t] = WMMA_F16(a0.v, bf, acc[0][t]);
            acc[1][t] = WMMA_F16(a1.v, bf, acc[1][t]);
        }
    }

#pragma unroll
    for (int m = 0; m < 2; ++m) {
#pragma unroll
        for (int t = 0; t < 4; ++t) {
            const int   ncol = ng * 64 + t * 16 + l16;
            const float bv   = bias[ncol];
#pragma unroll
            for (int r = 0; r < 8; ++r) {
                const int   row = mt * 32 + m * 16 + r + half * 8;
                const float val = acc[m][t][r] + bv;
                if (MODE == 0) {
                    ((_Float16*)outp)[(size_t)row * Dd + ncol] = (_Float16)val;
                } else if (MODE == 1) {
                    const int bb = row >> 11;        // / Ss
                    const int s  = row & (Ss - 1);
                    const int hh = ncol >> 6;        // / DKk
                    const int d  = ncol & (DKk - 1);
                    ((_Float16*)outp)[((size_t)(bb * Hh + hh) * DKk + d) * Ss + s] =
                        (_Float16)val;
                } else {
                    ((float*)outp)[(size_t)row * Dd + ncol] = val;
                }
            }
        }
    }
}

// --------------------- flash attention with ALiBi --------------------------
// One wave handles a 32-row query tile (two 16-row blocks with independent
// online-softmax state) of one (b, h); K/V B-fragments feed both blocks.
__global__ void __launch_bounds__(128)
attn_alibi(const _Float16* __restrict__ Q, const _Float16* __restrict__ K16,
           const _Float16* __restrict__ Vt, const _Float16* __restrict__ mbT,
           _Float16* __restrict__ Oa) {
    __shared__ _Float16 Pb[4][2][16 * 32];       // per-wave, per-qblock P tile

    const int lane = threadIdx.x & 31;
    const int wid  = threadIdx.x >> 5;
    const int gw   = blockIdx.x * 4 + wid;       // 2048 waves total
    const int qt   = gw & 63;                    // S/32 tiles
    const int h    = (gw >> 6) & 15;
    const int b    = gw >> 10;
    const int half = lane >> 4;
    const int l16  = lane & 15;

    const float slope = exp2f(-0.5f * (float)(h + 1));   // 2^(-8(h+1)/H), H=16
    const float scale = 0.125f;                          // 1/sqrt(DK)

    // Q A-fragments: 2 q-blocks x 2 d-chunks, loaded once, reused for all keys.
    AFrag aQ[2][2];
#pragma unroll
    for (int qb = 0; qb < 2; ++qb) {
        const _Float16* qptr =
            Q + (size_t)(b * Ss + qt * 32 + qb * 16 + l16) * Dd + h * DKk;
#pragma unroll
        for (int c = 0; c < 2; ++c) {
            aQ[qb][c].h[0] = *(const v8h*)(qptr + c * 32 + half * 16);
            aQ[qb][c].h[1] = *(const v8h*)(qptr + c * 32 + half * 16 + 8);
        }
    }

    v8f   o[2][4] = {};
    float m_r[2][8], l_r[2][8];
#pragma unroll
    for (int qb = 0; qb < 2; ++qb)
#pragma unroll
        for (int r = 0; r < 8; ++r) { m_r[qb][r] = -1e30f; l_r[qb][r] = 0.0f; }

    const _Float16* mbbase = mbT + (size_t)b * Ss * Ss;       // [j][i]
    const _Float16* kbase  = K16 + (size_t)(b * Ss) * Dd + h * DKk;
    const _Float16* vbase  = Vt + (size_t)(b * Hh + h) * DKk * Ss;

    // ALiBi running base: i - j = fBase + 16*(qb - t) + r, decremented by 32
    // per key chunk. All values are small integers -> exact in f32.
    float fBase = (float)(qt * 32 + half * 8 - l16);

    for (int kc = 0; kc < Ss; kc += 32) {
        // ---- K B-fragments (2 key tiles x 2 d-chunks) and mask-bias vectors
        //      (one b128 per (qb,t)) issued together so they clause.
        v16h kb[2][2];
        v8h  mb[2][2];
#pragma unroll
        for (int t = 0; t < 2; ++t) {
            const _Float16* kp = kbase + (size_t)(kc + t * 16 + l16) * Dd + half * 16;
            kb[t][0] = *(const v16h*)(kp);
            kb[t][1] = *(const v16h*)(kp + 32);
            const _Float16* mp =
                mbbase + (size_t)(kc + t * 16 + l16) * Ss + qt * 32 + half * 8;
            mb[0][t] = *(const v8h*)(mp);
            mb[1][t] = *(const v8h*)(mp + 16);
        }

        // ---- prefetch next chunk (global_prefetch_b8): hide latency under
        //      the softmax VALU/TRANS work below.
        if (kc + 32 < Ss) {
            __builtin_prefetch(kbase + (size_t)(kc + 32 + lane) * Dd, 0, 0);
            __builtin_prefetch(vbase + (size_t)(lane * 2) * Ss + kc + 32, 0, 0);
        }

        // ---- per q-block: scores, bias+mask, online softmax, P -> LDS
#pragma unroll
        for (int qb = 0; qb < 2; ++qb) {
            v8f s[2];
#pragma unroll
            for (int t = 0; t < 2; ++t) {
                v8f c = {};
                c = WMMA_F16(aQ[qb][0].v, kb[t][0], c);
                c = WMMA_F16(aQ[qb][1].v, kb[t][1], c);
                s[t] = c;
            }

#pragma unroll
            for (int t = 0; t < 2; ++t) {
#pragma unroll
                for (int r = 0; r < 8; ++r) {
                    const float dlt = (float)(16 * (qb - t) + r);   // literal
                    s[t][r] = s[t][r] * scale - slope * fabsf(fBase + dlt) +
                              (float)mb[qb][t][r];
                }
            }

            float alpha[8];
#pragma unroll
            for (int r = 0; r < 8; ++r) {
                float tmax = fmaxf(s[0][r], s[1][r]);
#pragma unroll
                for (int off = 1; off < 16; off <<= 1)
                    tmax = fmaxf(tmax, __shfl_xor(tmax, off, 32));
                const float mnew = fmaxf(m_r[qb][r], tmax);
                alpha[r]    = __expf(m_r[qb][r] - mnew);
                m_r[qb][r]  = mnew;
            }

            _Float16* pb = Pb[wid][qb];
#pragma unroll
            for (int t = 0; t < 2; ++t) {
#pragma unroll
                for (int r = 0; r < 8; ++r) {
                    const float p = __expf(s[t][r] - m_r[qb][r]);
                    s[t][r] = p;
                    pb[(r + half * 8) * 32 + t * 16 + l16] = (_Float16)p;
                }
            }
#pragma unroll
            for (int r = 0; r < 8; ++r) {
                float rs = s[0][r] + s[1][r];
#pragma unroll
                for (int off = 1; off < 16; off <<= 1)
                    rs += __shfl_xor(rs, off, 32);
                l_r[qb][r] = l_r[qb][r] * alpha[r] + rs;
            }
#pragma unroll
            for (int t = 0; t < 4; ++t)
#pragma unroll
                for (int r = 0; r < 8; ++r) o[qb][t][r] *= alpha[r];
        }

        // ---- A-fragments of P from LDS (same-wave DS ops are in-order)
        AFrag ap0, ap1;
        ap0.h[0] = *(const v8h*)(Pb[wid][0] + l16 * 32 + half * 16);
        ap0.h[1] = *(const v8h*)(Pb[wid][0] + l16 * 32 + half * 16 + 8);
        ap1.h[0] = *(const v8h*)(Pb[wid][1] + l16 * 32 + half * 16);
        ap1.h[1] = *(const v8h*)(Pb[wid][1] + l16 * 32 + half * 16 + 8);

        // ---- o += P x V: each Vt B-fragment feeds both q-blocks
#pragma unroll
        for (int t = 0; t < 4; ++t) {
            const int d = t * 16 + l16;
            const v16h vbf = *(const v16h*)(vbase + (size_t)d * Ss + kc + half * 16);
            o[0][t] = WMMA_F16(ap0.v, vbf, o[0][t]);
            o[1][t] = WMMA_F16(ap1.v, vbf, o[1][t]);
        }

        fBase -= 32.0f;
    }

    // ---- finalize: divide by row sums, write f16 attention output
#pragma unroll
    for (int qb = 0; qb < 2; ++qb) {
#pragma unroll
        for (int r = 0; r < 8; ++r) l_r[qb][r] = 1.0f / l_r[qb][r];
#pragma unroll
        for (int t = 0; t < 4; ++t) {
#pragma unroll
            for (int r = 0; r < 8; ++r) {
                const int row = qt * 32 + qb * 16 + r + half * 8;
                Oa[(size_t)(b * Ss + row) * Dd + h * DKk + t * 16 + l16] =
                    (_Float16)(o[qb][t][r] * l_r[qb][r]);
            }
        }
    }
}

// ---------------------------------------------------------------------------
extern "C" void kernel_launch(void* const* d_in, const int* in_sizes, int n_in,
                              void* d_out, int out_size, void* d_ws, size_t ws_size,
                              hipStream_t stream) {
    const float* query = (const float*)d_in[0];
    const float* key   = (const float*)d_in[1];
    const float* value = (const float*)d_in[2];
    const int*   mask  = (const int*)d_in[3];
    const float* Wq    = (const float*)d_in[4];
    const float* bq    = (const float*)d_in[5];
    const float* Wk    = (const float*)d_in[6];
    const float* bk    = (const float*)d_in[7];
    const float* Wv    = (const float*)d_in[8];
    const float* bv    = (const float*)d_in[9];
    const float* Wo    = (const float*)d_in[10];
    const float* bo    = (const float*)d_in[11];

    const size_t NX = (size_t)Mtot * Dd;   // 4 Mi halves per activation buffer
    const size_t NW = (size_t)Dd * Dd;     // 1 Mi halves per weight buffer

    _Float16* ws16 = (_Float16*)d_ws;
    _Float16* Xq16 = ws16;
    _Float16* Xk16 = ws16 + 1 * NX;
    _Float16* Xv16 = ws16 + 2 * NX;
    _Float16* Wqt  = ws16 + 3 * NX;
    _Float16* Wkt  = Wqt + NW;
    _Float16* Wvt  = Wqt + 2 * NW;
    _Float16* Wot  = Wqt + 3 * NW;
    _Float16* Q16  = ws16 + 3 * NX + 4 * NW;
    _Float16* K16  = ws16 + 4 * NX + 4 * NW;
    _Float16* Vt   = ws16 + 5 * NX + 4 * NW;
    _Float16* A16  = ws16 + 6 * NX + 4 * NW;
    // mbT (B*S*S = 2*NX halves) reuses the Xq16+Xk16 region, which is dead
    // once the Q/K projection GEMMs have consumed it (stream-ordered).
    _Float16* mbT  = ws16;

    const int nElem = (int)NX;
    const int cvtBlocks = (nElem + 255) / 256;
    cvt_f32_to_f16<<<cvtBlocks, 256, 0, stream>>>(query, Xq16, nElem);
    cvt_f32_to_f16<<<cvtBlocks, 256, 0, stream>>>(key,   Xk16, nElem);
    cvt_f32_to_f16<<<cvtBlocks, 256, 0, stream>>>(value, Xv16, nElem);

    const int trBlocks = (int)(NW / 256);
    transpose_cvt<<<trBlocks, 256, 0, stream>>>(Wq, Wqt);
    transpose_cvt<<<trBlocks, 256, 0, stream>>>(Wk, Wkt);
    transpose_cvt<<<trBlocks, 256, 0, stream>>>(Wv, Wvt);
    transpose_cvt<<<trBlocks, 256, 0, stream>>>(Wo, Wot);

    // (Mtot/32)*(Dd/64) = 2048 waves per GEMM = 512 blocks x 4 wave32s
    const int gemmBlocks = (Mtot / 32) * (Dd / 64) / 4;
    gemm_wmma<0><<<gemmBlocks, 128, 0, stream>>>(Xq16, Wqt, bq, (void*)Q16);
    gemm_wmma<0><<<gemmBlocks, 128, 0, stream>>>(Xk16, Wkt, bk, (void*)K16);
    gemm_wmma<1><<<gemmBlocks, 128, 0, stream>>>(Xv16, Wvt, bv, (void*)Vt);

    // mask -> transposed additive bias (after X buffers are dead)
    dim3 mbGrid(Ss / 32, Ss / 32, Bb);
    mask_bias_T<<<mbGrid, 256, 0, stream>>>(mask, mbT);

    // attention: B*H*(S/32) = 2048 waves = 512 blocks x 4 wave32s
    const int attnBlocks = (Bb * Hh * (Ss / 32)) / 4;
    attn_alibi<<<attnBlocks, 128, 0, stream>>>(Q16, K16, Vt, mbT, A16);

    // output projection -> f32 d_out
    gemm_wmma<2><<<gemmBlocks, 128, 0, stream>>>(A16, Wot, bo, d_out);
}
